// RGCN_28252294873236
// MI455X (gfx1250) — matrix operand order repaired and checked by reading, not compiled, not run on hardware
//
#include <hip/hip_runtime.h>
#include <hip/hip_bf16.h>

// ---------------- types ----------------
typedef __bf16 bf16_t;
typedef bf16_t bvec8  __attribute__((ext_vector_type(8)));
typedef bf16_t bvec16 __attribute__((ext_vector_type(16)));
typedef float  fvec8  __attribute__((ext_vector_type(8)));

#define DIM 128
#define LDSW 136    // 128 + 8 bf16 pad -> 272B row stride, 16B aligned, bank-conflict free
#define STRIPS 4    // 64-node strips computed per W staging

__device__ __forceinline__ float bf2f(unsigned short b) {
    union { unsigned int u; float f; } v; v.u = ((unsigned int)b) << 16; return v.f;
}
__device__ __forceinline__ bvec16 cat16(bvec8 lo, bvec8 hi) {
    bvec16 r;
#pragma unroll
    for (int i = 0; i < 8; ++i) { r[i] = lo[i]; r[i + 8] = hi[i]; }
    return r;
}

// ------------------------------------------------------------------
// Dense transform: out[r] = act(X) @ W[r]  (bf16 WMMA, fp32 accumulate)
// Block: 128 threads (4 waves). Each block stages W[r] once (bf16,
// transposed in LDS), then computes STRIPS x (64 rows x 128 cols).
// BF16OUT=1: write bf16 message table xw[r,N,128] (rel = blockIdx.y).
// BF16OUT=0: write fp32 out (self-loop path) + bias.
// RELU_IN : apply ReLU to X while staging (fuses layer-1 activation).
// ------------------------------------------------------------------
template<bool BF16OUT, bool RELU_IN>
__global__ __launch_bounds__(128) void rgcn_transform(
    const float* __restrict__ X, const float* __restrict__ Wg,
    bf16_t* __restrict__ outBf, float* __restrict__ outF,
    const float* __restrict__ bias, int nNodes)
{
    __shared__ __align__(16) bf16_t sW[DIM * LDSW];   // W[r]^T : [h][k]
    __shared__ __align__(16) bf16_t sA[64 * LDSW];    // A tile : [row][k]

    const int tid    = threadIdx.x;
    const int rel    = blockIdx.y;
    const int block0 = blockIdx.x * (64 * STRIPS);
    const float* Wr  = Wg + (size_t)rel * DIM * DIM;

    // Stage W[r] (fp32 [k][h]) -> sW bf16 transposed [h][k]  (once per block)
    for (int idx = tid; idx < DIM * 32; idx += 128) {
        int d = idx >> 5, c4 = idx & 31;
        float4 w = ((const float4*)(Wr + (size_t)d * DIM))[c4];
        int h0 = c4 * 4;
        sW[(h0 + 0) * LDSW + d] = (bf16_t)w.x;
        sW[(h0 + 1) * LDSW + d] = (bf16_t)w.y;
        sW[(h0 + 2) * LDSW + d] = (bf16_t)w.z;
        sW[(h0 + 3) * LDSW + d] = (bf16_t)w.w;
    }

    const int wave = tid >> 5, lane = tid & 31;
    const int hl = lane & 15, ksel = lane >> 4;

    for (int sIdx = 0; sIdx < STRIPS; ++sIdx) {
        const int node0 = block0 + sIdx * 64;
        if (node0 >= nNodes) break;                       // block-uniform

        // Stage A tile (64 rows of X) -> sA bf16 (optional fused ReLU)
        for (int idx = tid; idx < 64 * 32; idx += 128) {
            int rl = idx >> 5, c4 = idx & 31;
            int node = node0 + rl;
            float4 v = make_float4(0.f, 0.f, 0.f, 0.f);
            if (node < nNodes) v = ((const float4*)(X + (size_t)node * DIM))[c4];
            if (RELU_IN) {
                v.x = fmaxf(v.x, 0.f); v.y = fmaxf(v.y, 0.f);
                v.z = fmaxf(v.z, 0.f); v.w = fmaxf(v.w, 0.f);
            }
            bf16_t* p = sA + rl * LDSW + c4 * 4;
            p[0] = (bf16_t)v.x; p[1] = (bf16_t)v.y;
            p[2] = (bf16_t)v.z; p[3] = (bf16_t)v.w;
        }
        __syncthreads();   // A (and W on first iter) visible to all waves

        // A fragments for 4 K-steps, reused across 8 col tiles.
        // A lane layout: elems 0..7 -> K = 8*ksel+i ; 8..15 -> K = 16+8*ksel+i
        bvec16 afrag[4];
#pragma unroll
        for (int kk = 0; kk < 4; ++kk) {
            const bf16_t* b = sA + (wave * 16 + hl) * LDSW + kk * 32 + ksel * 8;
            afrag[kk] = cat16(*(const bvec8*)(b), *(const bvec8*)(b + 16));
        }
        __syncthreads();   // afrags in registers; sA may be overwritten next iter

        const bool fullTile = (node0 + 64 <= nNodes);
#pragma unroll
        for (int nt = 0; nt < 8; ++nt) {
            fvec8 acc = {};
#pragma unroll
            for (int kk = 0; kk < 4; ++kk) {
                // B lane layout: elems 0..7 -> K = 8*ksel+i ; 8..15 -> K = 16+8*ksel+i, col = lane&15
                const bf16_t* b = sW + (nt * 16 + hl) * LDSW + kk * 32 + ksel * 8;
                bvec16 bfrag = cat16(*(const bvec8*)(b), *(const bvec8*)(b + 16));
                acc = __builtin_amdgcn_wmma_f32_16x16x32_bf16(
                    false, afrag[kk], false, bfrag, (short)0, acc, false, false);
            }
            const int col = nt * 16 + hl;
            const int rbase = wave * 16 + 8 * ksel;       // first row this lane owns
            if (BF16OUT) {
                bf16_t* ob = outBf + ((size_t)rel * nNodes + node0 + rbase) * DIM + col;
                if (fullTile) {
#pragma unroll
                    for (int i = 0; i < 8; ++i) ob[(size_t)i * DIM] = (bf16_t)acc[i];
                } else {
#pragma unroll
                    for (int i = 0; i < 8; ++i)
                        if (node0 + rbase + i < nNodes) ob[(size_t)i * DIM] = (bf16_t)acc[i];
                }
            } else {
                const float bv = bias[col];
                float* ob = outF + (size_t)(node0 + rbase) * DIM + col;
                if (fullTile) {
#pragma unroll
                    for (int i = 0; i < 8; ++i) ob[(size_t)i * DIM] = acc[i] + bv;
                } else {
#pragma unroll
                    for (int i = 0; i < 8; ++i)
                        if (node0 + rbase + i < nNodes) ob[(size_t)i * DIM] = acc[i] + bv;
                }
            }
        }
    }
}

// ------------------------------------------------------------------
// Edge scatter: one wave32 per edge, grid-stride. Scalarized (SMEM)
// index loads; gather 128-wide bf16 message xw[etype, src]; 4 hardware
// float atomics per lane into acc[dst]. Next edge's message row is
// software-prefetched (global_prefetch_b8) while atomics are in flight.
// ------------------------------------------------------------------
__global__ __launch_bounds__(256) void rgcn_edge_scatter(
    const bf16_t* __restrict__ xw, const int* __restrict__ src,
    const int* __restrict__ dst, const int* __restrict__ et,
    float* __restrict__ acc, int nNodes, int nEdges)
{
    const int lane = threadIdx.x & 31;
    const int stride = gridDim.x * 8;
    int e = blockIdx.x * 8 + (threadIdx.x >> 5);
    if (e >= nEdges) return;
    e = __builtin_amdgcn_readfirstlane(e);                // scalarize index chain
    int s = src[e], d = dst[e], r = et[e];
    for (;;) {
        const int en = e + stride;
        const bool more = en < nEdges;
        int sn = 0, dn = 0, rn = 0;
        if (more) {
            sn = src[en]; dn = dst[en]; rn = et[en];
            __builtin_prefetch(xw + ((size_t)rn * nNodes + sn) * DIM + lane * 4, 0, 3);
        }
        const unsigned short* mp =
            (const unsigned short*)xw + ((size_t)r * nNodes + s) * DIM + lane * 4;
        uint2 m = *(const uint2*)mp;                      // 4 bf16 per lane (b64)
        float* o = acc + (size_t)d * DIM + lane * 4;
        unsafeAtomicAdd(o + 0, bf2f((unsigned short)(m.x & 0xFFFFu)));
        unsafeAtomicAdd(o + 1, bf2f((unsigned short)(m.x >> 16)));
        unsafeAtomicAdd(o + 2, bf2f((unsigned short)(m.y & 0xFFFFu)));
        unsafeAtomicAdd(o + 3, bf2f((unsigned short)(m.y >> 16)));
        if (!more) break;
        e = en; s = sn; d = dn; r = rn;
    }
}

extern "C" void kernel_launch(void* const* d_in, const int* in_sizes, int n_in,
                              void* d_out, int out_size, void* d_ws, size_t ws_size,
                              hipStream_t stream) {
    const float* features = (const float*)d_in[0];
    const float* W1    = (const float*)d_in[1];
    const float* loop1 = (const float*)d_in[2];
    const float* b1    = (const float*)d_in[3];
    const float* W2    = (const float*)d_in[4];
    const float* loop2 = (const float*)d_in[5];
    const float* b2    = (const float*)d_in[6];
    const int* src = (const int*)d_in[7];
    const int* dst = (const int*)d_in[8];
    const int* et  = (const int*)d_in[9];

    const int nNodes = in_sizes[0] / DIM;
    const int nEdges = in_sizes[7];
    const int nRels  = in_sizes[1] / (DIM * DIM);

    // Workspace: xw (bf16, [R,N,128]) then h (fp32, [N,128])
    bf16_t* xw = (bf16_t*)d_ws;
    size_t xwBytes = ((size_t)nRels * nNodes * DIM * sizeof(bf16_t) + 255) & ~(size_t)255;
    float* hbuf = (float*)((char*)d_ws + xwBytes);

    const int nodesPerBlock = 64 * STRIPS;
    dim3 tgridR((nNodes + nodesPerBlock - 1) / nodesPerBlock, nRels);
    dim3 tgrid1((nNodes + nodesPerBlock - 1) / nodesPerBlock, 1);
    int egrid = 4096;                                     // grid-stride edge waves
    if (egrid * 8 > nEdges) egrid = (nEdges + 7) / 8;

    // ---- Layer 1 : h = x@loop1 + b1 (init), += scatter(W1 x_src)
    rgcn_transform<true,  false><<<tgridR, 128, 0, stream>>>(features, W1, xw, nullptr, nullptr, nNodes);
    rgcn_transform<false, false><<<tgrid1, 128, 0, stream>>>(features, loop1, nullptr, hbuf, b1, nNodes);
    rgcn_edge_scatter<<<egrid, 256, 0, stream>>>(xw, src, dst, et, hbuf, nNodes, nEdges);

    // ---- Layer 2 : ReLU fused into A-staging of both transforms
    rgcn_transform<true,  true><<<tgridR, 128, 0, stream>>>(hbuf, W2, xw, nullptr, nullptr, nNodes);
    rgcn_transform<false, true><<<tgrid1, 128, 0, stream>>>(hbuf, loop2, nullptr, (float*)d_out, b2, nNodes);
    rgcn_edge_scatter<<<egrid, 256, 0, stream>>>(xw, src, dst, et, (float*)d_out, nNodes, nEdges);
}
